// BlockdiagButterflyLinear_1185410974366
// MI455X (gfx1250) — matrix operand order repaired and checked by reading, not compiled, not run on hardware
//
#include <hip/hip_runtime.h>

// ---------------------------------------------------------------------------
// BlockdiagButterflyLinear on MI455X (gfx1250, wave32, WMMA bf16)
//   Stage 1: tmp[b,k,j,y] = sum_i x[b, k*1024+i] * w1[k, j*48+y, i]
//   Stage 2: out[b, l*4+j] = sum_{k,y} tmp[b,k,j,y] * w2[j, l, k*48+y] + bias
// f32 -> bf16 conversion on the fly, f32 accumulation via
// v_wmma_f32_16x16x32_bf16.  Double-buffered LDS pipeline; bf16 operands are
// moved with global_load_async_to_lds_b128 (ASYNCcnt-tracked).
// ---------------------------------------------------------------------------

typedef __bf16   v16bf __attribute__((ext_vector_type(16)));
typedef __bf16   v8bf  __attribute__((ext_vector_type(8)));
typedef __bf16   v2bf  __attribute__((ext_vector_type(2)));
typedef float    v8f   __attribute__((ext_vector_type(8)));
typedef float    f32x4 __attribute__((ext_vector_type(4)));
typedef int      v4i   __attribute__((ext_vector_type(4)));
typedef unsigned int u32x2 __attribute__((ext_vector_type(2)));
typedef unsigned int u32x4 __attribute__((ext_vector_type(4)));

#define LDS_STRIDE 40   // halves per row: 32 data + 8 pad (80B, 16B-aligned)

#if __has_builtin(__builtin_amdgcn_global_load_async_to_lds_b128)
#define HAS_ASYNC 1
#else
#define HAS_ASYNC 0
#endif

// 16-byte global -> LDS copy. Async (ASYNCcnt-tracked, no VGPR staging) when
// the toolchain exposes the gfx1250 builtin; plain copy otherwise.
static __device__ __forceinline__ void cp16_g2l(const unsigned short* g,
                                                unsigned short* l) {
#if HAS_ASYNC
    __builtin_amdgcn_global_load_async_to_lds_b128(
        (__attribute__((address_space(1))) v4i*)(unsigned short*)g,
        (__attribute__((address_space(3))) v4i*)l,
        /*offset=*/0, /*cpol=*/0);
#else
    *(u32x4*)l = *(const u32x4*)g;
#endif
}

static __device__ __forceinline__ void wait_async0() {
#if HAS_ASYNC
 #if __has_builtin(__builtin_amdgcn_s_wait_asynccnt)
    __builtin_amdgcn_s_wait_asynccnt(0);
 #else
    asm volatile("s_wait_asynccnt 0" ::: "memory");
 #endif
#endif
}

// Native fptrunc: lowers to hardware bf16 convert if gfx1250 has one,
// otherwise clang emits an RNE software sequence.
static __device__ __forceinline__ unsigned short f2bf(float f) {
    return __builtin_bit_cast(unsigned short, (__bf16)f);
}
static __device__ __forceinline__ unsigned int pkbf(float a, float b) {
    v2bf r;
    r.x = (__bf16)a;
    r.y = (__bf16)b;
    return __builtin_bit_cast(unsigned int, r);
}

// A fragment, 16x32 bf16 tile starting at LDS row `rowBase`.
// lanes 0-15: row=lane,    K chunks at 0 and 16
// lanes16-31: row=lane-16, K chunks at 8 and 24
static __device__ __forceinline__ v16bf load_frag_A(const unsigned short* Xs,
                                                    int rowBase, int lane) {
    int lo  = (lane >> 4) * 8;
    int row = rowBase + (lane & 15);
    const unsigned short* p = Xs + row * LDS_STRIDE + lo;
    v8bf a0 = *(const v8bf*)(p);
    v8bf a1 = *(const v8bf*)(p + 16);
    return __builtin_shufflevector(a0, a1, 0,1,2,3,4,5,6,7,8,9,10,11,12,13,14,15);
}

// B fragment, 32x16 bf16 tile; LDS holds [col][K] (weight rows are B columns).
// lanes 0-15: col=lane, K chunks at 0 and 16; lanes 16-31: col=lane-16, K chunks at 8 and 24
static __device__ __forceinline__ v16bf load_frag_B(const unsigned short* Ws,
                                                    int colBase, int lane) {
    int col = colBase + (lane & 15);
    int ko  = (lane >> 4) * 8;
    const unsigned short* p = Ws + col * LDS_STRIDE + ko;
    v8bf b0 = *(const v8bf*)(p);
    v8bf b1 = *(const v8bf*)(p + 16);
    return __builtin_shufflevector(b0, b1, 0,1,2,3,4,5,6,7,8,9,10,11,12,13,14,15);
}

static __device__ __forceinline__ v8f wmma_bf16(v16bf a, v16bf b, v8f c) {
    return __builtin_amdgcn_wmma_f32_16x16x32_bf16(
        false, a, false, b, (short)0, c, false, false);
}

// ---------------------------------------------------------------------------
// Kernel 0: convert f32 weights to bf16 in scratch (packed, 2 elems/iter)
// ---------------------------------------------------------------------------
__global__ __launch_bounds__(256)
void cvt_weights(const float* __restrict__ w1, const float* __restrict__ w2,
                 unsigned int* __restrict__ w1b, unsigned int* __restrict__ w2b,
                 int npairs) {
    int i = blockIdx.x * blockDim.x + threadIdx.x;
    if (i < npairs) {
        w1b[i] = pkbf(w1[2 * i], w1[2 * i + 1]);
        w2b[i] = pkbf(w2[2 * i], w2[2 * i + 1]);
    }
}

// ---------------------------------------------------------------------------
// Kernel 1: stage-1 GEMM.  grid = (B/128, 192/64, 4), 256 thr (8 waves)
//   X_k (B x 1024, f32)  x  W1_k^T (1024 x 192, bf16)  ->  tmp (bf16)
//   tmp layout: [b][j][k][y]  (addr = b*768 + j*192 + k*48 + y)
// Double-buffered: X via register-staged pipeline (needs f32->bf16 VALU),
// W via async-to-LDS.
// ---------------------------------------------------------------------------
__global__ __launch_bounds__(256)
void gemm_stage1(const float* __restrict__ x,
                 const unsigned short* __restrict__ w1b,
                 unsigned short* __restrict__ tmp) {
    __shared__ __align__(16) unsigned short Xs[2][128 * LDS_STRIDE];
    __shared__ __align__(16) unsigned short Ws[2][64 * LDS_STRIDE];

    const int t     = threadIdx.x;
    const int lane  = t & 31;
    const int wid   = t >> 5;
    const int wm    = wid & 3;       // 4 waves along M (32 rows each)
    const int wn    = wid >> 2;      // 2 waves along N (32 cols each)
    const int mBase = blockIdx.x * 128;
    const int nBase = blockIdx.y * 64;
    const int k     = blockIdx.z;

    f32x4 xr[4];
    auto loadX = [&](int kb) {                       // global f32 loads only
        #pragma unroll
        for (int it = 0; it < 4; ++it) {
            int idx = t + it * 256, r = idx >> 3, c4 = idx & 7;
            xr[it] = *(const f32x4*)(x + (size_t)(mBase + r) * 4096
                                       + (size_t)k * 1024 + kb * 32 + c4 * 4);
        }
    };
    auto storeX = [&](int buf) {                     // convert + LDS commit
        #pragma unroll
        for (int it = 0; it < 4; ++it) {
            int idx = t + it * 256, r = idx >> 3, c4 = idx & 7;
            u32x2 pk;
            pk.x = pkbf(xr[it].x, xr[it].y);
            pk.y = pkbf(xr[it].z, xr[it].w);
            *(u32x2*)&Xs[buf][r * LDS_STRIDE + c4 * 4] = pk;
        }
    };
    auto copyW = [&](int kb, int buf) {              // 64 cols x 32 bf16
        int col = t >> 2, c8 = t & 3;
        cp16_g2l(w1b + ((size_t)(k * 192 + nBase + col)) * 1024 + kb * 32 + c8 * 8,
                 &Ws[buf][col * LDS_STRIDE + c8 * 8]);
    };

    loadX(0);
    copyW(0, 0);
    storeX(0);
    wait_async0();
    __syncthreads();

    v8f acc[2][2] = {};
    for (int kb = 0; kb < 32; ++kb) {
        const int cur = kb & 1, nxt = kb + 1, nbuf = nxt & 1;
        if (nxt < 32) { loadX(nxt); copyW(nxt, nbuf); }   // issue ahead

        v16bf a0 = load_frag_A(Xs[cur], wm * 32,      lane);
        v16bf a1 = load_frag_A(Xs[cur], wm * 32 + 16, lane);
        v16bf b0 = load_frag_B(Ws[cur], wn * 32,      lane);
        v16bf b1 = load_frag_B(Ws[cur], wn * 32 + 16, lane);
        acc[0][0] = wmma_bf16(a0, b0, acc[0][0]);
        acc[0][1] = wmma_bf16(a0, b1, acc[0][1]);
        acc[1][0] = wmma_bf16(a1, b0, acc[1][0]);
        acc[1][1] = wmma_bf16(a1, b1, acc[1][1]);

        if (nxt < 32) storeX(nbuf);                        // commit next tile
        wait_async0();
        __syncthreads();
    }

    // epilogue: scatter bf16 tmp in [b][j][k][y] layout
    for (int tm = 0; tm < 2; ++tm) {
        int rbase = mBase + wm * 32 + tm * 16 + (lane >> 4) * 8;
        for (int tn = 0; tn < 2; ++tn) {
            int colG = nBase + wn * 32 + tn * 16 + (lane & 15);  // 0..191
            int jj   = colG / 48;
            int y    = colG - jj * 48;
            size_t base = (size_t)jj * 192 + (size_t)k * 48 + y;
            #pragma unroll
            for (int v = 0; v < 8; ++v)
                tmp[(size_t)(rbase + v) * 768 + base] = f2bf(acc[tm][tn][v]);
        }
    }
}

// ---------------------------------------------------------------------------
// Kernel 2: stage-2 GEMM.  grid = (B/128, 1024/128, 4), 256 thr (8 waves)
//   T_j (B x 192, bf16)  x  W2_j^T (192 x 1024, bf16)  ->  out (f32) + bias
// WG tile 128x128, wave tile 32x64 (8 WMMAs / K-step).  Both operands bf16
// -> async-to-LDS double buffering, no VGPR staging.
// out column for local col l is l*4 + j (interleaved), row stride 4096.
// ---------------------------------------------------------------------------
__global__ __launch_bounds__(256)
void gemm_stage2(const unsigned short* __restrict__ tmp,
                 const unsigned short* __restrict__ w2b,
                 const float* __restrict__ bias,
                 float* __restrict__ out) {
    __shared__ __align__(16) unsigned short As[2][128 * LDS_STRIDE];
    __shared__ __align__(16) unsigned short Ws[2][128 * LDS_STRIDE];

    const int t     = threadIdx.x;
    const int lane  = t & 31;
    const int wid   = t >> 5;
    const int wm    = wid & 3;       // 4 waves along M (32 rows each)
    const int wn    = wid >> 2;      // 2 waves along N (64 cols each)
    const int mBase = blockIdx.x * 128;
    const int nBase = blockIdx.y * 128;   // 0..896
    const int j     = blockIdx.z;

    auto copyA = [&](int kb, int buf) {              // 128 rows x 32 bf16
        #pragma unroll
        for (int it = 0; it < 2; ++it) {
            int idx = t + it * 256, r = idx >> 2, c8 = idx & 3;
            cp16_g2l(tmp + (size_t)(mBase + r) * 768 + (size_t)j * 192
                         + kb * 32 + c8 * 8,
                     &As[buf][r * LDS_STRIDE + c8 * 8]);
        }
    };
    auto copyW = [&](int kb, int buf) {              // 128 cols x 32 bf16
        #pragma unroll
        for (int it = 0; it < 2; ++it) {
            int idx = t + it * 256, col = idx >> 2, c8 = idx & 3;
            cp16_g2l(w2b + ((size_t)(j * 1024 + nBase + col)) * 192
                         + kb * 32 + c8 * 8,
                     &Ws[buf][col * LDS_STRIDE + c8 * 8]);
        }
    };

    copyA(0, 0);
    copyW(0, 0);
    wait_async0();
    __syncthreads();

    v8f acc[2][4] = {};
    for (int kb = 0; kb < 6; ++kb) {                 // K = 192 = 6 x 32
        const int cur = kb & 1, nxt = kb + 1, nbuf = nxt & 1;
        if (nxt < 6) { copyA(nxt, nbuf); copyW(nxt, nbuf); }

        v16bf a0 = load_frag_A(As[cur], wm * 32,      lane);
        v16bf a1 = load_frag_A(As[cur], wm * 32 + 16, lane);
        #pragma unroll
        for (int tn = 0; tn < 4; ++tn) {
            v16bf b = load_frag_B(Ws[cur], wn * 64 + tn * 16, lane);
            acc[0][tn] = wmma_bf16(a0, b, acc[0][tn]);
            acc[1][tn] = wmma_bf16(a1, b, acc[1][tn]);
        }

        wait_async0();
        __syncthreads();
    }

    // epilogue: out[b, l*4+j] = acc + bias
    for (int tm = 0; tm < 2; ++tm) {
        int rbase = mBase + wm * 32 + tm * 16 + (lane >> 4) * 8;
        for (int tn = 0; tn < 4; ++tn) {
            int l      = nBase + wn * 64 + tn * 16 + (lane & 15);  // 0..1023
            int outCol = l * 4 + j;
            float bv   = bias[outCol];
            #pragma unroll
            for (int v = 0; v < 8; ++v)
                out[(size_t)(rbase + v) * 4096 + outCol] = acc[tm][tn][v] + bv;
        }
    }
}

// ---------------------------------------------------------------------------
extern "C" void kernel_launch(void* const* d_in, const int* in_sizes, int n_in,
                              void* d_out, int out_size, void* d_ws, size_t ws_size,
                              hipStream_t stream) {
    const float* x    = (const float*)d_in[0];   // (4, 2048, 4096) f32
    const float* w1   = (const float*)d_in[1];   // (4, 192, 1024) f32
    const float* w2   = (const float*)d_in[2];   // (4, 1024, 192) f32
    const float* bias = (const float*)d_in[3];   // (4096,) f32
    float* out        = (float*)d_out;

    const int B  = in_sizes[0] / 4096;           // 8192 flattened rows
    const int NW = 4 * 192 * 1024;               // 786432 elems per weight tensor

    unsigned short* ws  = (unsigned short*)d_ws;
    unsigned short* w1b = ws;                    // 1.5 MB
    unsigned short* w2b = ws + NW;               // 1.5 MB
    unsigned short* tmp = ws + 2 * NW;           // B*768 bf16 = 12.6 MB (L2)

    cvt_weights<<<(NW / 2 + 255) / 256, 256, 0, stream>>>(
        w1, w2, (unsigned int*)w1b, (unsigned int*)w2b, NW / 2);

    dim3 grid1(B / 128, 192 / 64, 4);
    gemm_stage1<<<grid1, 256, 0, stream>>>(x, w1b, tmp);

    dim3 grid2(B / 128, 1024 / 128, 4);
    gemm_stage2<<<grid2, 256, 0, stream>>>(tmp, w2b, bias, out);
}
